// MHSA_72086731096207
// MI455X (gfx1250) — matrix operand order repaired and checked by reading, not compiled, not run on hardware
//
#include <hip/hip_runtime.h>
#include <hip/hip_bf16.h>

#define DIM 1024
#define HEADS 16
#define HEAD_DIM 64
#define BATCH 4
#define SEQ 2048
#define NROWS (BATCH * SEQ) // 8192

typedef __attribute__((ext_vector_type(16))) __bf16 bf16x16;
typedef __attribute__((ext_vector_type(8)))  float  f32x8;

union FragB16 {
    bf16x16 v;
    uint4   q[2];
    unsigned short u[16];
};

__device__ __forceinline__ unsigned short f32_to_bf16(float f) {
    union { float f; unsigned int u; } c; c.f = f;
    unsigned int r = c.u + 0x7FFFu + ((c.u >> 16) & 1u); // round to nearest even
    return (unsigned short)(r >> 16);
}

// ---------------------------------------------------------------------------
// Kernel 1: LayerNorm over D=1024, emit bf16 rows
// ---------------------------------------------------------------------------
__global__ __launch_bounds__(256)
void ln_bf16_kernel(const float* __restrict__ x,
                    const float* __restrict__ gamma,
                    const float* __restrict__ beta,
                    unsigned short* __restrict__ h) {
    const int row = blockIdx.x;
    const float* xr = x + (size_t)row * DIM;

    __shared__ float red[2][8];

    float v[4];
    float s = 0.f, s2 = 0.f;
#pragma unroll
    for (int i = 0; i < 4; ++i) {
        v[i] = xr[threadIdx.x + i * 256];
        s += v[i];
        s2 += v[i] * v[i];
    }
#pragma unroll
    for (int m = 16; m >= 1; m >>= 1) {
        s  += __shfl_xor(s, m);
        s2 += __shfl_xor(s2, m);
    }
    const int wave = threadIdx.x >> 5;
    if ((threadIdx.x & 31) == 0) { red[0][wave] = s; red[1][wave] = s2; }
    __syncthreads();
    s = 0.f; s2 = 0.f;
#pragma unroll
    for (int w = 0; w < 8; ++w) { s += red[0][w]; s2 += red[1][w]; }

    const float mu   = s * (1.0f / DIM);
    const float var  = s2 * (1.0f / DIM) - mu * mu;
    const float rstd = rsqrtf(var + 1e-5f);

    unsigned short* hr = h + (size_t)row * DIM;
#pragma unroll
    for (int i = 0; i < 4; ++i) {
        const int c = threadIdx.x + i * 256;
        hr[c] = f32_to_bf16((v[i] - mu) * rstd * gamma[c] + beta[c]);
    }
}

// ---------------------------------------------------------------------------
// Kernel 2: fp32 -> bf16 conversion (weights)
// ---------------------------------------------------------------------------
__global__ void f32_to_bf16_kernel(const float* __restrict__ src,
                                   unsigned short* __restrict__ dst, int n) {
    const int i = blockIdx.x * blockDim.x + threadIdx.x;
    if (i < n) dst[i] = f32_to_bf16(src[i]);
}

// ---------------------------------------------------------------------------
// Kernel 3/5: C[M,N] = A[M,K] * Bw[N,K]^T  (bf16 in, fp32 accum via WMMA)
// Block tile 128x256, 8 waves (2 rows x 4 cols), each wave 64x64:
// 4x4 grid of 16x16 accumulators, K-step 32 -> 16 WMMAs per wave per step.
// ---------------------------------------------------------------------------
template <bool OUT_BF16>
__global__ __launch_bounds__(256)
void gemm_bf16_wmma(const unsigned short* __restrict__ A,
                    const unsigned short* __restrict__ Bw,
                    void* __restrict__ C,
                    int M, int N, int K) {
    __shared__ __align__(16) unsigned short As[128 * 32];
    __shared__ __align__(16) unsigned short Bs[256 * 32];

    const int m0 = blockIdx.y * 128;
    const int n0 = blockIdx.x * 256;
    const int t = threadIdx.x;
    const int wave = t >> 5;
    const int lane = t & 31;
    const int laneHi = (lane >> 4) & 1;
    const int ln = lane & 15;
    const int wm = (wave >> 2) * 64; // 2 wave-rows
    const int wn = (wave & 3) * 64;  // 4 wave-cols

    f32x8 acc[4][4] = {};

    for (int k0 = 0; k0 < K; k0 += 32) {
        __syncthreads();
        // stage A (128x32) and B (256x32): 16B chunks
#pragma unroll
        for (int i = 0; i < 2; ++i) {
            const int c = t + i * 256;          // 512 chunks
            const int row = c >> 2;
            const int seg = (c & 3) * 8;
            *(uint4*)(&As[row * 32 + seg]) =
                *(const uint4*)(&A[(size_t)(m0 + row) * K + k0 + seg]);
            if (k0 + 32 < K)
                __builtin_prefetch(&A[(size_t)(m0 + row) * K + k0 + 32 + seg], 0, 0);
        }
#pragma unroll
        for (int i = 0; i < 4; ++i) {
            const int c = t + i * 256;          // 1024 chunks
            const int row = c >> 2;
            const int seg = (c & 3) * 8;
            *(uint4*)(&Bs[row * 32 + seg]) =
                *(const uint4*)(&Bw[(size_t)(n0 + row) * K + k0 + seg]);
            if (k0 + 32 < K)
                __builtin_prefetch(&Bw[(size_t)(n0 + row) * K + k0 + 32 + seg], 0, 0);
        }
        __syncthreads();

        // A fragments (16x32): lane<16 row=ln K={0..7,16..23}; lane>=16 K={8..15,24..31}
        FragB16 af[4];
#pragma unroll
        for (int i = 0; i < 4; ++i) {
            const unsigned short* p = &As[(wm + i * 16 + ln) * 32 + laneHi * 8];
            af[i].q[0] = *(const uint4*)(p);
            af[i].q[1] = *(const uint4*)(p + 16);
        }
        // B fragments (32x16): n=lane%16, lane<16 K={0..7,16..23}; lane>=16 K={8..15,24..31}
        FragB16 bfr[4];
#pragma unroll
        for (int j = 0; j < 4; ++j) {
            const unsigned short* p = &Bs[(wn + j * 16 + ln) * 32 + laneHi * 8];
            bfr[j].q[0] = *(const uint4*)(p);
            bfr[j].q[1] = *(const uint4*)(p + 16);
        }
#pragma unroll
        for (int i = 0; i < 4; ++i)
#pragma unroll
            for (int j = 0; j < 4; ++j)
                acc[i][j] = __builtin_amdgcn_wmma_f32_16x16x32_bf16(
                    false, af[i].v, false, bfr[j].v, (short)0, acc[i][j], false, false);
    }

#pragma unroll
    for (int i = 0; i < 4; ++i)
#pragma unroll
        for (int j = 0; j < 4; ++j)
#pragma unroll
            for (int r = 0; r < 8; ++r) {
                const int m = m0 + wm + i * 16 + r + laneHi * 8;
                const int n = n0 + wn + j * 16 + ln;
                const float val = acc[i][j][r];
                if (OUT_BF16)
                    ((unsigned short*)C)[(size_t)m * N + n] = f32_to_bf16(val);
                else
                    ((float*)C)[(size_t)m * N + n] = val;
            }
}

// ---------------------------------------------------------------------------
// Kernel 4: flash attention. One block (4 waves) per (b, h, 64-query tile).
// qkv: [NROWS, 3*DIM] bf16.  o: [NROWS, DIM] bf16.
// Softmax in base-2: logits pre-scaled by log2(e)/sqrt(hd).
// ---------------------------------------------------------------------------
__global__ __launch_bounds__(128)
void attn_kernel(const unsigned short* __restrict__ qkv,
                 unsigned short* __restrict__ o) {
    const int qt = blockIdx.x;  // 0..31
    const int h  = blockIdx.y;  // 0..15
    const int b  = blockIdx.z;  // 0..3
    const int s0 = qt * 64;
    const size_t rowbase = (size_t)b * SEQ;
    const size_t ld = 3 * DIM;
    const float SCALE = 0.18033688011112042f; // (1/sqrt(64)) * log2(e)

    __shared__ __align__(16) unsigned short Qs[64 * 64];
    __shared__ __align__(16) unsigned short Ks[64 * 64];
    __shared__ __align__(16) unsigned short Vt[64 * 64]; // [d][t] transposed
    __shared__ __align__(16) unsigned short Ps[4][16 * 64];

    const int t = threadIdx.x;
    const int wave = t >> 5;
    const int lane = t & 31;
    const int laneHi = (lane >> 4) & 1;
    const int ln = lane & 15;

    const int qcol = h * HEAD_DIM;
    const int kcol = DIM + h * HEAD_DIM;
    const int vcol = 2 * DIM + h * HEAD_DIM;

    // Load Q tile 64x64
#pragma unroll
    for (int i = 0; i < 4; ++i) {
        const int c = t + i * 128;   // 512 chunks of 8 elems
        const int row = c >> 3;
        const int seg = (c & 7) * 8;
        *(uint4*)(&Qs[row * 64 + seg]) =
            *(const uint4*)(&qkv[(rowbase + s0 + row) * ld + qcol + seg]);
    }

    f32x8 oacc[4] = {};
    float mrow[8], lrow[8];
#pragma unroll
    for (int r = 0; r < 8; ++r) { mrow[r] = -3.0e38f; lrow[r] = 0.f; }

    for (int kt = 0; kt < SEQ / 64; ++kt) {
        __syncthreads(); // previous iteration's K/V/P consumers are done
#pragma unroll
        for (int i = 0; i < 4; ++i) {
            const int c = t + i * 128;
            const int row = c >> 3;
            const int seg = (c & 7) * 8;
            *(uint4*)(&Ks[row * 64 + seg]) =
                *(const uint4*)(&qkv[(rowbase + kt * 64 + row) * ld + kcol + seg]);
            const unsigned short* vp =
                &qkv[(rowbase + kt * 64 + row) * ld + vcol + seg];
#pragma unroll
            for (int j = 0; j < 8; ++j) Vt[(seg + j) * 64 + row] = vp[j];
        }
        __syncthreads();

        // S = Q @ K^T for this wave's 16 query rows x 64 keys
        f32x8 sacc[4] = {};
#pragma unroll
        for (int ks = 0; ks < 2; ++ks) {
            FragB16 aq;
            const unsigned short* ap = &Qs[(wave * 16 + ln) * 64 + ks * 32 + laneHi * 8];
            aq.q[0] = *(const uint4*)(ap);
            aq.q[1] = *(const uint4*)(ap + 16);
#pragma unroll
            for (int nt = 0; nt < 4; ++nt) {
                FragB16 bk;
                const unsigned short* bp = &Ks[(nt * 16 + ln) * 64 + ks * 32 + laneHi * 8];
                bk.q[0] = *(const uint4*)(bp);
                bk.q[1] = *(const uint4*)(bp + 16);
                sacc[nt] = __builtin_amdgcn_wmma_f32_16x16x32_bf16(
                    false, aq.v, false, bk.v, (short)0, sacc[nt], false, false);
            }
        }

        // Online softmax (base 2). C layout: row = r + 8*laneHi,
        // col = nt*16 + lane%16; a row lives in one 16-lane group.
#pragma unroll
        for (int r = 0; r < 8; ++r) {
            float smax = -3.0e38f;
#pragma unroll
            for (int nt = 0; nt < 4; ++nt) {
                sacc[nt][r] *= SCALE;
                smax = fmaxf(smax, sacc[nt][r]);
            }
#pragma unroll
            for (int m = 8; m >= 1; m >>= 1) smax = fmaxf(smax, __shfl_xor(smax, m));
            const float mnew  = fmaxf(mrow[r], smax);
            const float scale = exp2f(mrow[r] - mnew);
            const int   mloc  = r + laneHi * 8;
            float psum = 0.f;
#pragma unroll
            for (int nt = 0; nt < 4; ++nt) {
                const float p = exp2f(sacc[nt][r] - mnew);
                psum += p;
                Ps[wave][mloc * 64 + nt * 16 + ln] = f32_to_bf16(p);
            }
#pragma unroll
            for (int m = 8; m >= 1; m >>= 1) psum += __shfl_xor(psum, m);
            lrow[r] = lrow[r] * scale + psum;
            mrow[r] = mnew;
#pragma unroll
            for (int nt = 0; nt < 4; ++nt) oacc[nt][r] *= scale;
        }

        // O += P @ V  (A = Ps[wave] 16x64, B = Vt as [d][t], K-contiguous)
#pragma unroll
        for (int ks = 0; ks < 2; ++ks) {
            FragB16 apf;
            const unsigned short* ap = &Ps[wave][ln * 64 + ks * 32 + laneHi * 8];
            apf.q[0] = *(const uint4*)(ap);
            apf.q[1] = *(const uint4*)(ap + 16);
#pragma unroll
            for (int nt = 0; nt < 4; ++nt) {
                FragB16 bv;
                const unsigned short* bp = &Vt[(nt * 16 + ln) * 64 + ks * 32 + laneHi * 8];
                bv.q[0] = *(const uint4*)(bp);
                bv.q[1] = *(const uint4*)(bp + 16);
                oacc[nt] = __builtin_amdgcn_wmma_f32_16x16x32_bf16(
                    false, apf.v, false, bv.v, (short)0, oacc[nt], false, false);
            }
        }
    }

    // Normalize and emit bf16 head output
#pragma unroll
    for (int r = 0; r < 8; ++r) {
        const float inv = 1.0f / lrow[r];
        const int m = s0 + wave * 16 + r + laneHi * 8;
#pragma unroll
        for (int nt = 0; nt < 4; ++nt)
            o[(rowbase + m) * DIM + h * HEAD_DIM + nt * 16 + ln] =
                f32_to_bf16(oacc[nt][r] * inv);
    }
}

// ---------------------------------------------------------------------------
// Launch
// ---------------------------------------------------------------------------
extern "C" void kernel_launch(void* const* d_in, const int* in_sizes, int n_in,
                              void* d_out, int out_size, void* d_ws, size_t ws_size,
                              hipStream_t stream) {
    const float* x          = (const float*)d_in[0];
    const float* in_proj_w  = (const float*)d_in[1];
    const float* out_proj_w = (const float*)d_in[2];
    const float* ln_gamma   = (const float*)d_in[3];
    const float* ln_beta    = (const float*)d_in[4];
    float* out = (float*)d_out;

    unsigned short* ws   = (unsigned short*)d_ws;
    unsigned short* h    = ws;                                  // NROWS*DIM
    unsigned short* qkv  = h   + (size_t)NROWS * DIM;           // NROWS*3*DIM
    unsigned short* ob   = qkv + (size_t)NROWS * 3 * DIM;       // NROWS*DIM
    unsigned short* wIn  = ob  + (size_t)NROWS * DIM;           // 3*DIM*DIM
    unsigned short* wOut = wIn + (size_t)3 * DIM * DIM;         // DIM*DIM

    // 1. LayerNorm -> bf16
    ln_bf16_kernel<<<NROWS, 256, 0, stream>>>(x, ln_gamma, ln_beta, h);

    // 2. weights -> bf16
    {
        int n = 3 * DIM * DIM;
        f32_to_bf16_kernel<<<(n + 255) / 256, 256, 0, stream>>>(in_proj_w, wIn, n);
        n = DIM * DIM;
        f32_to_bf16_kernel<<<(n + 255) / 256, 256, 0, stream>>>(out_proj_w, wOut, n);
    }

    // 3. QKV projection: [8192,1024] x [3072,1024]^T -> [8192,3072] bf16
    {
        dim3 grid(3 * DIM / 256, NROWS / 128); // (12, 64)
        gemm_bf16_wmma<true><<<grid, 256, 0, stream>>>(h, wIn, qkv, NROWS, 3 * DIM, DIM);
    }

    // 4. Flash attention per (b, h, 64-query tile)
    {
        dim3 grid(SEQ / 64, HEADS, BATCH); // (32, 16, 4)
        attn_kernel<<<grid, 128, 0, stream>>>(qkv, ob);
    }

    // 5. Output projection: [8192,1024] x [1024,1024]^T -> fp32 d_out
    {
        dim3 grid(DIM / 256, NROWS / 128); // (4, 64)
        gemm_bf16_wmma<false><<<grid, 256, 0, stream>>>(ob, wOut, out, NROWS, DIM, DIM);
    }
}
